// GeneralizedAttention_23682449670147
// MI455X (gfx1250) — matrix operand order repaired and checked by reading, not compiled, not run on hardware
//
#include <hip/hip_runtime.h>
#include <hip/hip_bf16.h>

#define NBATCH   4
#define NSEQ     256
#define D_MODEL  256
#define NHEAD    8
#define HEAD_DIM 32
#define CHUNK    32     // j-rows per LDS chunk
#define NCHUNK   (NSEQ / CHUNK)

typedef __attribute__((ext_vector_type(16))) __bf16 v16bf;
typedef __attribute__((ext_vector_type(8)))  float  v8f;

union Frag16 { uint4 u[2]; v16bf v; };

__device__ __forceinline__ unsigned short f2bf_rne(float f) {
  union { float f; unsigned int u; } c; c.f = f;
  unsigned int u = c.u;
  return (unsigned short)((u + 0x7FFFu + ((u >> 16) & 1u)) >> 16);
}

// ---------------- W fp32 -> bf16 (row-major [dout][k]) ----------------
__global__ void ga_cvt_w_bf16(const float* __restrict__ W,
                              unsigned short* __restrict__ Wbf) {
  int idx = blockIdx.x * blockDim.x + threadIdx.x;
  Wbf[idx] = f2bf_rne(W[idx]);
}

// ---------------- y[row,:] = x[row,:] @ W^T + b (small GEMMs) ----------------
__global__ __launch_bounds__(256)
void ga_proj(const float* __restrict__ x, const float* __restrict__ W,
             const float* __restrict__ bias, float* __restrict__ y) {
  __shared__ float xr[D_MODEL];
  const int row = blockIdx.x;
  const int o   = threadIdx.x;
  xr[o] = x[(size_t)row * D_MODEL + o];
  __syncthreads();
  const float* w = W + (size_t)o * D_MODEL;
  float acc = bias[o];
  #pragma unroll 8
  for (int k = 0; k < D_MODEL; ++k) acc += xr[k] * w[k];
  y[(size_t)row * D_MODEL + o] = acc;
}

// ---------------- fused edge-proj + scores + softmax + AV ----------------
// grid = (NSEQ, NBATCH), block = 256 (8 waves); one block per (b,i)
__global__ __launch_bounds__(256)
void ga_attn(const float* __restrict__ edges,
             const float* __restrict__ node_p,         // fp32, includes bias
             const unsigned short* __restrict__ Wbf,   // bf16 W [dout][k]
             const float* __restrict__ bias,
             float* __restrict__ attn_out,             // [B,H,N,N]
             float* __restrict__ out_pre) {            // [B,N,D]
  extern __shared__ char smem[];
  unsigned short* Ebf = (unsigned short*)smem;                          // [CHUNK][256] bf16
  float* P  = (float*)(smem + CHUNK*D_MODEL*2);                         // [CHUNK][256] f32
  float* S  = (float*)(smem + CHUNK*D_MODEL*2 + CHUNK*D_MODEL*4);       // [H][N]
  float* NQ = (float*)(smem + CHUNK*D_MODEL*2 + CHUNK*D_MODEL*4 + NHEAD*NSEQ*4); // [256]

  const int tid  = threadIdx.x;
  const int wave = tid >> 5;
  const int lane = tid & 31;
  const int i = blockIdx.x;
  const int b = blockIdx.y;

  NQ[tid] = node_p[((size_t)(b*NSEQ + i))*D_MODEL + tid];

  const float* erow = edges + ((size_t)(b*NSEQ + i))*NSEQ*D_MODEL;

  // wave tile assignment: 2 jtiles x 16 ntiles = 32 tiles, 4 per wave
  const int jt  = wave >> 2;          // jtile (0..1)
  const int nt0 = (wave & 3) * 4;     // first ntile
  const int lh  = lane >> 4;          // lane half (0/1)
  const int ll  = lane & 15;

  // ---- preload ALL B fragments (bf16 W^T) into registers: 8 K-steps x 4 tiles ----
  // B fragment layout (32x16, 16-bit): lanes 0-15 hold K {0..7,16..23}, lanes 16-31 hold K {8..15,24..31}
  Frag16 bw[D_MODEL / 32][4];
  #pragma unroll
  for (int kb = 0; kb < D_MODEL / 32; ++kb) {
    #pragma unroll
    for (int r = 0; r < 4; ++r) {
      const unsigned short* bbase =
          Wbf + ((size_t)((nt0 + r)*16 + ll)) * D_MODEL + kb*32 + (lh << 3);
      bw[kb][r].u[0] = *(const uint4*)(bbase);
      bw[kb][r].u[1] = *(const uint4*)(bbase + 16);
    }
  }

  // ---- register double-buffer for the streamed edge chunk (8 x float4 / thread) ----
  const float4* srcbase = (const float4*)erow;     // CHUNK*D_MODEL/4 = 2048 float4 per chunk
  float4 vbuf[8];
  #pragma unroll
  for (int it = 0; it < 8; ++it) vbuf[it] = srcbase[(size_t)tid + 256*it];

  for (int chunk = 0; chunk < NCHUNK; ++chunk) {
    // ---- convert current chunk (already in registers) -> bf16 LDS tile ----
    #pragma unroll
    for (int it = 0; it < 8; ++it) {
      const int idx = tid + 256 * it;
      const float4 v = vbuf[it];
      uint2 pk;
      pk.x = (unsigned)f2bf_rne(v.x) | ((unsigned)f2bf_rne(v.y) << 16);
      pk.y = (unsigned)f2bf_rne(v.z) | ((unsigned)f2bf_rne(v.w) << 16);
      *(uint2*)(Ebf + idx * 4) = pk;
    }
    // ---- issue next chunk's global loads now; latency hidden behind GEMM+scores ----
    if (chunk + 1 < NCHUNK) {
      const float4* nsrc = srcbase + (size_t)(chunk + 1) * (CHUNK * D_MODEL / 4);
      #pragma unroll
      for (int it = 0; it < 8; ++it) vbuf[it] = nsrc[(size_t)tid + 256*it];
    }
    __syncthreads();

    // ---- P[jl, dout] = E @ W^T via v_wmma_f32_16x16x32_bf16 (A double-buffered) ----
    v8f zero = {};
    v8f acc[4] = {zero, zero, zero, zero};

    // A fragment layout (16x32, 16-bit): lanes 0-15: K {0..7,16..23}; lanes 16-31: K {8..15,24..31}
    const unsigned short* arow = Ebf + (jt*16 + ll) * D_MODEL + (lh << 3);
    Frag16 afc, afn;
    afc.u[0] = *(const uint4*)(arow);
    afc.u[1] = *(const uint4*)(arow + 16);

    #pragma unroll
    for (int kb = 0; kb < D_MODEL / 32; ++kb) {
      if (kb + 1 < D_MODEL / 32) {       // prefetch next A fragment from LDS
        const unsigned short* an = arow + (kb + 1) * 32;
        afn.u[0] = *(const uint4*)(an);
        afn.u[1] = *(const uint4*)(an + 16);
      }
      #pragma unroll
      for (int r = 0; r < 4; ++r) {
        acc[r] = __builtin_amdgcn_wmma_f32_16x16x32_bf16(
            false, afc.v, false, bw[kb][r].v, (short)0, acc[r], false, false);
      }
      afc = afn;
    }

    // ---- write projected tile + bias into LDS (D layout: M = v + 8*lh, N = ll) ----
    #pragma unroll
    for (int r = 0; r < 4; ++r) {
      const int ncol = (nt0 + r) * 16 + ll;
      const float bv = bias[ncol];
      #pragma unroll
      for (int v = 0; v < 8; ++v) {
        const int jl = jt * 16 + v + (lh << 3);
        P[jl * D_MODEL + ncol] = acc[r][v] + bv;
      }
    }
    __syncthreads();

    // ---- scores: s[h,j] = sum_d (e+q_i)·(e+k_j), one (h, j_local) per thread ----
    {
      const int h  = tid >> 5;           // 0..7
      const int jl = tid & 31;           // 0..31
      const int j  = chunk * CHUNK + jl;
      const float* pr = P  + jl * D_MODEL + h * HEAD_DIM;
      const float* nq = NQ + h * HEAD_DIM;
      const float* nk = node_p + ((size_t)(b*NSEQ + j))*D_MODEL + h*HEAD_DIM;
      float s = 0.f;
      #pragma unroll
      for (int d = 0; d < HEAD_DIM; ++d) {
        const float e = pr[d];
        s += (e + nq[d]) * (e + nk[d]);
      }
      S[h * NSEQ + j] = s;
    }
    __syncthreads();
  }

  // ---- tanh clip + softmax + AV: wave w owns head h = w ----
  const int h = wave;
  float* srow = S + h * NSEQ;
  const float scale = 0.17677669529663687f;   // 1/sqrt(32)
  float vloc[8];
  float m = -3.4e38f;
  #pragma unroll
  for (int t = 0; t < 8; ++t) {
    float x = 10.f * tanhf(srow[lane + 32*t] * scale);
    vloc[t] = x; m = fmaxf(m, x);
  }
  #pragma unroll
  for (int off = 16; off > 0; off >>= 1) m = fmaxf(m, __shfl_xor(m, off, 32));
  float sum = 0.f;
  #pragma unroll
  for (int t = 0; t < 8; ++t) { float e = __expf(vloc[t] - m); vloc[t] = e; sum += e; }
  #pragma unroll
  for (int off = 16; off > 0; off >>= 1) sum += __shfl_xor(sum, off, 32);
  const float rinv = 1.f / sum;

  float* arow = attn_out + (((size_t)b*NHEAD + h)*NSEQ + i)*NSEQ;
  #pragma unroll
  for (int t = 0; t < 8; ++t) {
    const float a = vloc[t] * rinv;
    srow[lane + 32*t] = a;        // wave-local reuse
    arow[lane + 32*t] = a;        // attn output
  }

  // out_pre[b,i,h*32+lane] = sum_j attn[h,j] * node_p[b,j,h*32+lane]
  const float* npb = node_p + (size_t)b*NSEQ*D_MODEL + h*HEAD_DIM + lane;
  float o = 0.f;
  for (int j = 0; j < NSEQ; ++j) o += srow[j] * npb[(size_t)j * D_MODEL];
  out_pre[((size_t)(b*NSEQ + i))*D_MODEL + h*HEAD_DIM + lane] = o;
}

extern "C" void kernel_launch(void* const* d_in, const int* in_sizes, int n_in,
                              void* d_out, int out_size, void* d_ws, size_t ws_size,
                              hipStream_t stream) {
  const float* nodes = (const float*)d_in[0];   // [B,N,D]
  const float* edges = (const float*)d_in[1];   // [B,N,N,D]
  const float* W     = (const float*)d_in[2];   // [D,D]
  const float* bias  = (const float*)d_in[3];   // [D]

  float* out  = (float*)d_out;                       // [B,N,D]
  float* attn = out + (size_t)NBATCH*NSEQ*D_MODEL;   // [B,H,N,N]

  char* ws = (char*)d_ws;
  unsigned short* Wbf = (unsigned short*)ws;                               // 128 KB
  float* node_p  = (float*)(ws + 131072);                                  // 1 MB
  float* out_pre = (float*)(ws + 131072 + (size_t)NBATCH*NSEQ*D_MODEL*4);  // 1 MB

  // 1) W -> bf16
  ga_cvt_w_bf16<<<(D_MODEL*D_MODEL)/256, 256, 0, stream>>>(W, Wbf);
  // 2) node projection (includes bias)
  ga_proj<<<NBATCH*NSEQ, 256, 0, stream>>>(nodes, W, bias, node_p);
  // 3) fused edge-proj + scores + softmax + AV
  const size_t smem = (size_t)CHUNK*D_MODEL*2     // E bf16
                    + (size_t)CHUNK*D_MODEL*4     // P f32
                    + (size_t)NHEAD*NSEQ*4        // scores
                    + (size_t)D_MODEL*4;          // NQ
  dim3 grid(NSEQ, NBATCH);
  ga_attn<<<grid, 256, smem, stream>>>(edges, node_p, Wbf, bias, attn, out_pre);
  // 4) final output projection
  ga_proj<<<NBATCH*NSEQ, 256, 0, stream>>>(out_pre, W, bias, out);
}
